// Rwkv6SelfAttention_53223234732569
// MI455X (gfx1250) — matrix operand order repaired and checked by reading, not compiled, not run on hardware
//
#include <hip/hip_runtime.h>
#include <math.h>
#include <stdint.h>

typedef __attribute__((ext_vector_type(16))) _Float16 v16h;
typedef __attribute__((ext_vector_type(8)))  float    v8f;

constexpr int TT  = 1024;
constexpr int HID = 2048;
constexpr int HH  = 32;
constexpr int HD  = 64;
constexpr int TMD = 32;   // TM
constexpr int TDD = 64;   // TD

enum { EPI_NONE = 0, EPI_TANH, EPI_SILU, EPI_MIX, EPI_DECAY, EPI_RESID };

#if __has_builtin(__builtin_amdgcn_global_load_async_to_lds_b32) && \
    __has_builtin(__builtin_amdgcn_s_wait_asynccnt)
#define HAS_ASYNC_LDS 1
#else
#define HAS_ASYNC_LDS 0
#endif

#if HAS_ASYNC_LDS
typedef __attribute__((address_space(1))) int g1int;
typedef __attribute__((address_space(3))) int l3int;
__device__ __forceinline__ void async_ld_f32(const float* g, const void* lds) {
  __builtin_amdgcn_global_load_async_to_lds_b32(
      (g1int*)(uintptr_t)g,
      (l3int*)(uint32_t)(uintptr_t)lds,
      0, 0);
}
#endif

// Fast activations: gfx1250 has a hardware v_tanh_f32 transcendental.
__device__ __forceinline__ float fast_tanh(float x) {
#if __has_builtin(__builtin_amdgcn_tanhf)
  return __builtin_amdgcn_tanhf(x);
#else
  return tanhf(x);
#endif
}
__device__ __forceinline__ float fast_exp(float x) { return __expf(x); }
__device__ __forceinline__ float fast_silu(float x) {
  return x * __frcp_rn(1.f + __expf(-x));
}

// ---------------------------------------------------------------------------
// Generic WMMA GEMM:  C[m,n] = epi( sum_k A[m,k] * Bk(n,k) )
//   BT==1 : B is (N x K) row-major (C = A @ B^T) -> direct contiguous loads
//   BT==0 : B is (K x N) row-major -> wave-cooperative LDS transpose staging
// Each wave computes one 16-row M tile x NT 16-col N tiles: the A fragment is
// converted once per k-step and reused by NT back-to-back WMMAs.
// Fragment packing: A and B both use contiguous K = 16*hi + i (16-bit A 16x32, B 32x16, f32 C/D).
// ---------------------------------------------------------------------------
template<int BT, int EPI, int NT>
__global__ void wmma_gemm(const float* __restrict__ A, int lda,
                          const float* __restrict__ B, int ldb,
                          float* __restrict__ C, int ldc,
                          int M, int N, int K,
                          const float* __restrict__ e0,   // xl (MIX) / residual (RESID)
                          const float* __restrict__ e1,   // sx (MIX)
                          const float* __restrict__ e2)   // time_maa row (MIX) / time_decay (DECAY)
{
  const int wave = (int)((blockIdx.x * blockDim.x + threadIdx.x) >> 5);
  const int wib  = (int)(threadIdx.x >> 5);   // wave index within block (0..7)
  const int lane = (int)(threadIdx.x & 31);
  const int ng    = N / (16 * NT);
  const int tiles = (M >> 4) * ng;
  if (wave >= tiles) return;                 // wave-uniform: EXEC stays all-ones
  const int tm  = (wave / ng) << 4;
  const int tn0 = (wave % ng) * (16 * NT);
  const int hi  = lane >> 4;                 // lane half (0: lanes 0-15, 1: 16-31)
  const int l   = lane & 15;

  // LDS staging for the BT==0 transpose: [n][k], k-row padded to 40 halves
  // (80 B) so the 16-half fragment reads are 16B-aligned ds_load_b128s.
  constexpr int BN = BT ? 1 : 16 * NT;
  constexpr int BK = BT ? 1 : 40;
  __shared__ _Float16 bst[8][BN][BK];

  v8f acc[NT] = {};
  const float* arow = A + (size_t)(tm + l) * lda;

  for (int k0 = 0; k0 < K; k0 += 32) {
    // A 16x32 f16 fragment: lane half selects K base 0/16; K = 16*hi + {0..15}.
    v16h af;
    const int ba = k0 + (hi << 4);
#pragma unroll
    for (int i = 0; i < 8; ++i)  af[i]     = (_Float16)arow[ba + i];
#pragma unroll
    for (int i = 0; i < 8; ++i)  af[8 + i] = (_Float16)arow[ba + 8 + i];

    if (!BT) {
      // Cooperative stage: lane owns k-row (k0+lane); coalesced row load,
      // f16 convert, transposed store into [n][k].
      const float* brow = B + (size_t)(k0 + lane) * ldb + tn0;
#pragma unroll
      for (int n = 0; n < 16 * NT; ++n)
        bst[wib][n][lane] = (_Float16)brow[n];
    }

#pragma unroll
    for (int nt = 0; nt < NT; ++nt) {
      // B 32x16 f16 fragment: n = tn + l, K = 16*hi + {0..15}.
      v16h bf;
      if (BT) {
        const float* bp = B + (size_t)(tn0 + nt * 16 + l) * ldb + k0 + (hi << 4);
        __builtin_prefetch(bp + 64, 0, 0);   // global_prefetch_b8
#pragma unroll
        for (int i = 0; i < 16; ++i) bf[i] = (_Float16)bp[i];
      } else {
        const _Float16* bp = &bst[wib][nt * 16 + l][hi << 4];
#pragma unroll
        for (int i = 0; i < 16; ++i) bf[i] = bp[i];
      }
      acc[nt] = __builtin_amdgcn_wmma_f32_16x16x32_f16(false, af, false, bf,
                                                       (short)0, acc[nt],
                                                       false, false);
    }
  }

  // C/D layout: lane half -> M base 0/8, VGPR j -> row offset j, col = tn+l.
#pragma unroll
  for (int nt = 0; nt < NT; ++nt) {
    const int cn = tn0 + nt * 16 + l;
#pragma unroll
    for (int j = 0; j < 8; ++j) {
      const int cm = tm + (hi << 3) + j;
      float r = acc[nt][j];
      if (EPI == EPI_TANH) r = fast_tanh(r);
      if (EPI == EPI_SILU) r = fast_silu(r);
      if (EPI == EPI_MIX) {
        const size_t ix = (size_t)cm * ldc + cn;
        r = e0[ix] + e1[ix] * (r + e2[cn]);
      }
      if (EPI == EPI_DECAY) {
        float td = e2[cn] + r;
        td = fminf(fmaxf(td, -9.72f), 2.27f);
        r = fast_exp(-fast_exp(td));
      }
      if (EPI == EPI_RESID) r = e0[(size_t)cm * ldc + cn] + r;
      C[(size_t)cm * ldc + cn] = r;
    }
  }
}

// ---------------------------------------------------------------------------
// LayerNorm over HID per token; also emits state1_out from the last token.
// ---------------------------------------------------------------------------
__global__ void ln1_kernel(const float* __restrict__ x,
                           const float* __restrict__ w,
                           const float* __restrict__ b,
                           float* __restrict__ xl,
                           float* __restrict__ state1_out)
{
  const int t   = blockIdx.x;
  const int tid = threadIdx.x;
  const float* xr = x + (size_t)t * HID;
  float sum = 0.f, ssq = 0.f;
  for (int i = tid; i < HID; i += 256) { float v = xr[i]; sum += v; ssq += v * v; }
  __shared__ float r1[256], r2[256];
  r1[tid] = sum; r2[tid] = ssq; __syncthreads();
  for (int off = 128; off > 0; off >>= 1) {
    if (tid < off) { r1[tid] += r1[tid + off]; r2[tid] += r2[tid + off]; }
    __syncthreads();
  }
  const float mean = r1[0] * (1.f / HID);
  const float var  = r2[0] * (1.f / HID) - mean * mean;
  const float inv  = rsqrtf(var + 1e-5f);
  float* out = xl + (size_t)t * HID;
  for (int i = tid; i < HID; i += 256) {
    float v = (xr[i] - mean) * inv * w[i] + b[i];
    out[i] = v;
    if (t == TT - 1) state1_out[i] = v;
  }
}

// sx = past - xl ; xxx_in = xl + sx * time_maa_x
__global__ void shift_kernel(const float* __restrict__ xl,
                             const float* __restrict__ state1,
                             const float* __restrict__ tmaax,
                             float* __restrict__ sx,
                             float* __restrict__ xxx_in)
{
  const int i = blockIdx.x * blockDim.x + threadIdx.x;
  if (i >= TT * HID) return;
  const int t = i >> 11;       // / HID
  const int c = i & (HID - 1);
  const float cur  = xl[i];
  const float prev = (t == 0) ? state1[c] : xl[i - HID];
  const float s = prev - cur;
  sx[i] = s;
  xxx_in[i] = cur + s * tmaax[c];
}

// ---------------------------------------------------------------------------
// Serial WKV scan: 1 block per head, 1 thread per output column e.
// 64x64 fp32 state lives in VGPRs. Per-step r/k/w vectors are double-buffered
// in LDS; the NEXT step's vectors are fetched with ASYNCcnt-tracked
// global_load_async_to_lds while the current step's 64 dependent FMAs run.
// ---------------------------------------------------------------------------
__global__ void wkv_scan(const float* __restrict__ r,
                         const float* __restrict__ k,
                         const float* __restrict__ v,
                         const float* __restrict__ w,
                         const float* __restrict__ time_first,
                         const float* __restrict__ state2_in,
                         float* __restrict__ wkv,
                         float* __restrict__ state2_out)
{
  const int h = blockIdx.x;    // head
  const int e = threadIdx.x;   // output column 0..63

  float s[HD];
#pragma unroll
  for (int d = 0; d < HD; ++d)
    s[d] = state2_in[((size_t)h * HD + d) * HD + e];

  __shared__ float sh_r[2][HD], sh_k[2][HD], sh_w[2][HD], sh_tf[HD];
  sh_tf[e] = time_first[h * HD + e];
  {
    const size_t i0 = (size_t)h * HD + e;   // t = 0
    sh_r[0][e] = r[i0];
    sh_k[0][e] = k[i0];
    sh_w[0][e] = w[i0];
  }
  __syncthreads();

  for (int t = 0; t < TT; ++t) {
    const int buf = t & 1;
    const size_t idx = ((size_t)t * HH + h) * HD + e;

#if HAS_ASYNC_LDS
    if (t + 1 < TT) {             // prefetch step t+1 into the other buffer
      const size_t nx = idx + (size_t)HH * HD;
      async_ld_f32(r + nx, &sh_r[buf ^ 1][e]);
      async_ld_f32(k + nx, &sh_k[buf ^ 1][e]);
      async_ld_f32(w + nx, &sh_w[buf ^ 1][e]);
    }
#endif

    const float ve = v[idx];
    float acc = 0.f;
#pragma unroll
    for (int d = 0; d < HD; ++d) {
      const float rd = sh_r[buf][d];
      const float kd = sh_k[buf][d];
      const float sd = s[d];
      acc  = fmaf(rd, fmaf(kd * sh_tf[d], ve, sd), acc);
      s[d] = fmaf(sd, sh_w[buf][d], kd * ve);
    }
    wkv[idx] = acc;

#if HAS_ASYNC_LDS
    if (t + 1 < TT) __builtin_amdgcn_s_wait_asynccnt(0);
    __syncthreads();
#else
    if (t + 1 < TT) {
      const size_t nx = idx + (size_t)HH * HD;
      __syncthreads();
      sh_r[buf ^ 1][e] = r[nx];
      sh_k[buf ^ 1][e] = k[nx];
      sh_w[buf ^ 1][e] = w[nx];
      __syncthreads();
    }
#endif
  }

#pragma unroll
  for (int d = 0; d < HD; ++d)
    state2_out[((size_t)h * HD + d) * HD + e] = s[d];
}

// GroupNorm over each head's 64 values + lnx affine + SiLU-gated multiply.
__global__ void gn_gate(const float* __restrict__ wkv,
                        const float* __restrict__ g,
                        const float* __restrict__ lnx_w,
                        const float* __restrict__ lnx_b,
                        float* __restrict__ xo)
{
  const int th = blockIdx.x;          // t*HH + h
  const int e  = threadIdx.x;         // 0..63
  const int h  = th & (HH - 1);
  const size_t base = (size_t)th * HD;
  const float val = wkv[base + e];
  __shared__ float s1[HD], s2[HD];
  s1[e] = val; s2[e] = val * val; __syncthreads();
  for (int off = 32; off > 0; off >>= 1) {
    if (e < off) { s1[e] += s1[e + off]; s2[e] += s2[e + off]; }
    __syncthreads();
  }
  const float mean = s1[0] * (1.f / HD);
  const float var  = s2[0] * (1.f / HD) - mean * mean;
  const float xn   = (val - mean) * rsqrtf(var + 1e-5f);
  const int  hd    = h * HD + e;
  xo[base + e] = (xn * lnx_w[hd] + lnx_b[hd]) * g[base + e];
}

// ---------------------------------------------------------------------------
extern "C" void kernel_launch(void* const* d_in, const int* in_sizes, int n_in,
                              void* d_out, int out_size, void* d_ws, size_t ws_size,
                              hipStream_t stream) {
  const float* x         = (const float*)d_in[0];
  const float* state1    = (const float*)d_in[1];
  const float* state2    = (const float*)d_in[2];
  const float* ln1_w     = (const float*)d_in[3];
  const float* ln1_b     = (const float*)d_in[4];
  const float* tmaax     = (const float*)d_in[5];
  const float* time_maa  = (const float*)d_in[6];
  const float* maa_w1    = (const float*)d_in[7];
  const float* maa_w2    = (const float*)d_in[8];
  const float* td_w1     = (const float*)d_in[9];
  const float* td_w2     = (const float*)d_in[10];
  const float* tdecay    = (const float*)d_in[11];
  const float* tfirst    = (const float*)d_in[12];
  const float* Wr        = (const float*)d_in[13];
  const float* Wk        = (const float*)d_in[14];
  const float* Wv        = (const float*)d_in[15];
  const float* Wg        = (const float*)d_in[16];
  const float* Wo        = (const float*)d_in[17];
  const float* lnx_w     = (const float*)d_in[18];
  const float* lnx_b     = (const float*)d_in[19];

  float* out = (float*)d_out;
  float* state1_out = out + (size_t)TT * HID;
  float* state2_out = state1_out + HID;

  // workspace layout (floats)
  float* ws = (float*)d_ws;
  const size_t S = (size_t)TT * HID;
  float* xl   = ws;
  float* sx   = ws + 1 * S;
  float* mix0 = ws + 2 * S;   // mw
  float* mix1 = ws + 3 * S;   // mk
  float* mix2 = ws + 4 * S;   // mv
  float* mix3 = ws + 5 * S;   // mr
  float* mix4 = ws + 6 * S;   // mg
  float* rb   = ws + 7 * S;
  float* kb   = ws + 8 * S;
  float* vb   = ws + 9 * S;
  float* gb   = ws + 10 * S;
  float* wb   = ws + 11 * S;  // decay w
  float* wkvb = ws + 12 * S;  // scan output (also reused as xxx_in earlier)
  float* xxx_in = wkvb;       // dead before scan writes wkv
  float* xxx160 = ws + 13 * S;               // T x 160
  float* tdh    = xxx160 + (size_t)TT * 160; // T x 64
  float* xo     = mix0;        // mw dead after td gemm1; reuse for gated output

  auto nblk = [](int M, int N, int NT) {
    return (((M >> 4) * (N / (16 * NT))) + 7) / 8;
  };

  ln1_kernel<<<TT, 256, 0, stream>>>(x, ln1_w, ln1_b, xl, state1_out);
  shift_kernel<<<(TT * HID) / 256, 256, 0, stream>>>(xl, state1, tmaax, sx, xxx_in);

  // xxx = tanh(xxx_in @ maa_w1)                    (T x 160)
  wmma_gemm<0, EPI_TANH, 2><<<nblk(TT, 5 * TMD, 2), 256, 0, stream>>>(
      xxx_in, HID, maa_w1, 5 * TMD, xxx160, 5 * TMD, TT, 5 * TMD, HID,
      nullptr, nullptr, nullptr);

  // mix_s = xl + sx * (xxx_s @ maa_w2[s] + time_maa[s])   for s in 0..4
  float* mixes[5] = { mix0, mix1, mix2, mix3, mix4 };
  for (int s = 0; s < 5; ++s)
    wmma_gemm<0, EPI_MIX, 2><<<nblk(TT, HID, 2), 256, 0, stream>>>(
        xxx160 + s * TMD, 5 * TMD, maa_w2 + (size_t)s * TMD * HID, HID,
        mixes[s], HID, TT, HID, TMD, xl, sx, time_maa + (size_t)s * HID);

  // td path: tanh(mw @ td_w1) @ td_w2 -> decay w
  wmma_gemm<0, EPI_TANH, 2><<<nblk(TT, TDD, 2), 256, 0, stream>>>(
      mix0, HID, td_w1, TDD, tdh, TDD, TT, TDD, HID, nullptr, nullptr, nullptr);
  wmma_gemm<0, EPI_DECAY, 2><<<nblk(TT, HID, 2), 256, 0, stream>>>(
      tdh, TDD, td_w2, HID, wb, HID, TT, HID, TDD, nullptr, nullptr, tdecay);

  // r / k / v / g projections (C = mix @ W^T)
  wmma_gemm<1, EPI_NONE, 4><<<nblk(TT, HID, 4), 256, 0, stream>>>(
      mix3, HID, Wr, HID, rb, HID, TT, HID, HID, nullptr, nullptr, nullptr);
  wmma_gemm<1, EPI_NONE, 4><<<nblk(TT, HID, 4), 256, 0, stream>>>(
      mix1, HID, Wk, HID, kb, HID, TT, HID, HID, nullptr, nullptr, nullptr);
  wmma_gemm<1, EPI_NONE, 4><<<nblk(TT, HID, 4), 256, 0, stream>>>(
      mix2, HID, Wv, HID, vb, HID, TT, HID, HID, nullptr, nullptr, nullptr);
  wmma_gemm<1, EPI_SILU, 4><<<nblk(TT, HID, 4), 256, 0, stream>>>(
      mix4, HID, Wg, HID, gb, HID, TT, HID, HID, nullptr, nullptr, nullptr);

  // serial WKV scan (state in VGPRs), emits wkv + state2_out
  wkv_scan<<<HH, HD, 0, stream>>>(rb, kb, vb, wb, tfirst, state2, wkvb, state2_out);

  // per-head GroupNorm + gate
  gn_gate<<<TT * HH, HD, 0, stream>>>(wkvb, gb, lnx_w, lnx_b, xo);

  // output projection with fused residual: out = x + xo @ Wo^T
  wmma_gemm<1, EPI_RESID, 4><<<nblk(TT, HID, 4), 256, 0, stream>>>(
      xo, HID, Wo, HID, out, HID, TT, HID, HID, x, nullptr, nullptr);
}
